// MAB_17592186044743
// MI455X (gfx1250) — matrix operand order repaired and checked by reading, not compiled, not run on hardware
//
#include <hip/hip_runtime.h>
#include <hip/hip_bf16.h>
#include <math.h>

// ---------------------------------------------------------------------------
// CDNA5 (gfx1250) WMMA types
// ---------------------------------------------------------------------------
typedef __attribute__((ext_vector_type(16))) __bf16 v16bf;
typedef __attribute__((ext_vector_type(8)))  __bf16 v8bf;
typedef __attribute__((ext_vector_type(8)))  float  v8f;
typedef __attribute__((ext_vector_type(4)))  unsigned u32x4;
typedef __attribute__((ext_vector_type(4)))  unsigned short u16x4;

typedef unsigned short bf16_t; // raw bf16 storage (host-safe)

static __device__ __forceinline__ bf16_t f2b(float f) {
  return __builtin_bit_cast(bf16_t, (__bf16)f); // v_cvt_pk_bf16_f32 capable
}
static __device__ __forceinline__ v8bf ld8bf(const bf16_t* p) {
  return __builtin_bit_cast(v8bf, *(const u32x4*)p); // 16B vector load
}

// ---------------------------------------------------------------------------
// Fragment loaders (CDNA5 wave32 WMMA 16x16x32 bf16 layouts, CDNA5 ISA)
// A (16x32): lane<16 -> row=lane,     K {0..7, 16..23}
//            lane>=16 -> row=lane-16, K {8..15, 24..31}
// B (32x16): lane<16 -> col=lane,     K {0..7, 16..23}
//            lane>=16 -> col=lane-16, K {8..15, 24..31}
// C (16x16 f32): lane&15 = N; vgpr i = row i (lane<16) / row 8+i (lane>=16)
// ---------------------------------------------------------------------------
static __device__ __forceinline__ v16bf load_A_frag_bf(const bf16_t* base, int ld,
                                                       int lane) {
  int row  = lane & 15;
  int kofs = (lane < 16) ? 0 : 8;
  const bf16_t* p = base + (size_t)row * ld + kofs;
  v8bf lo = ld8bf(p);
  v8bf hi = ld8bf(p + 16);
  return __builtin_shufflevector(lo, hi, 0, 1, 2, 3, 4, 5, 6, 7,
                                 8, 9, 10, 11, 12, 13, 14, 15);
}

// base points at B-column origin: element (k,n) lives at base[n*ld + k]
static __device__ __forceinline__ v16bf load_B_frag_bf(const bf16_t* base, int ld,
                                                       int lane) {
  int col  = lane & 15;
  int kofs = (lane < 16) ? 0 : 8;
  const bf16_t* p = base + (size_t)col * ld + kofs;
  v8bf lo = ld8bf(p);
  v8bf hi = ld8bf(p + 16);
  return __builtin_shufflevector(lo, hi, 0, 1, 2, 3, 4, 5, 6, 7,
                                 8, 9, 10, 11, 12, 13, 14, 15);
}

static __device__ __forceinline__ v8f wmma_bf16(v16bf a, v16bf b, v8f c) {
  return __builtin_amdgcn_wmma_f32_16x16x32_bf16(false, a, false, b, (short)0, c,
                                                 false, false);
}

// ---------------------------------------------------------------------------
// Elementwise f32 -> bf16 conversion (cold path; bandwidth-trivial)
// ---------------------------------------------------------------------------
__global__ __launch_bounds__(256)
void mab_cvt_bf16_kernel(const float* __restrict__ x, bf16_t* __restrict__ y, int n) {
  int idx = (blockIdx.x * 256 + threadIdx.x) * 4;
  if (idx < n) {
    const float4 v = *(const float4*)(x + idx);
    u16x4 r;
    r[0] = f2b(v.x); r[1] = f2b(v.y); r[2] = f2b(v.z); r[3] = f2b(v.w);
    *(u16x4*)(y + idx) = r;
  }
}

// ---------------------------------------------------------------------------
// WMMA GEMM: out = act( X[M,KK](bf16) @ W[NN,KK]^T(bf16) + bias ) (+residual)
// Block = 128 (4 waves); block tile 128(M) x 64(N); wave -> 32x64 (8 WMMA/step).
// Fully unrolled k-loop: loads of later steps overlap earlier WMMAs, no
// buffer-rotation movs. All strides compile-time.
// ---------------------------------------------------------------------------
template <int KK, int NN, int NPB>
__global__ __launch_bounds__(128)
void mab_gemm_kernel(const bf16_t* __restrict__ X, const bf16_t* __restrict__ W,
                     const float* __restrict__ bias,
                     const float* __restrict__ residual,   // may be null (post-act add)
                     float* __restrict__ outF,             // may be null
                     bf16_t* __restrict__ outB,            // may be null, [M,NN]
                     bf16_t* __restrict__ outBT,           // may be null, [Mb][NN][NPB]
                     int relu_flag) {
  const int lane = threadIdx.x & 31;
  const int wave = threadIdx.x >> 5;
  const int m0 = blockIdx.x * 128 + wave * 32;
  const int n0 = blockIdx.y * 64;

  v8f acc0[4] = {};
  v8f acc1[4] = {};

#pragma unroll
  for (int k0 = 0; k0 < KK; k0 += 32) {
    v16bf a0 = load_A_frag_bf(X + (size_t)m0 * KK + k0, KK, lane);
    v16bf a1 = load_A_frag_bf(X + (size_t)(m0 + 16) * KK + k0, KK, lane);
    v16bf b[4];
#pragma unroll
    for (int nt = 0; nt < 4; ++nt)
      b[nt] = load_B_frag_bf(W + (size_t)(n0 + nt * 16) * KK + k0, KK, lane);
#pragma unroll
    for (int nt = 0; nt < 4; ++nt) {
      acc0[nt] = wmma_bf16(a0, b[nt], acc0[nt]);
      acc1[nt] = wmma_bf16(a1, b[nt], acc1[nt]);
    }
  }

  const int colL   = lane & 15;
  const int rowAdd = (lane < 16) ? 0 : 8;
#pragma unroll
  for (int half = 0; half < 2; ++half) {
    v8f* acc = half ? acc1 : acc0;
#pragma unroll
    for (int nt = 0; nt < 4; ++nt) {
      int col = n0 + nt * 16 + colL;
      float bs = bias[col];
#pragma unroll
      for (int i = 0; i < 8; ++i) {
        int row = m0 + half * 16 + rowAdd + i;
        float v = acc[nt][i] + bs;
        if (relu_flag) v = fmaxf(v, 0.0f);
        size_t idx = (size_t)row * NN + col;
        if (residual) v += residual[idx];
        if (outF) outF[idx] = v;
        if (outB) outB[idx] = f2b(v);
        if (outBT) {
          int bb = row / NPB, nn = row - bb * NPB;
          outBT[((size_t)bb * NN + col) * NPB + nn] = f2b(v);
        }
      }
    }
  }
}

// ---------------------------------------------------------------------------
// Flash attention (per b, h, 64-query tile).  O = Qp + softmax(Qh Kh^T / 16) Vh
// Block = 128 (4 waves); wave -> 16-query strip; 64-key blocks (18 WMMA/step).
// Row-sum comes from a ones-column WMMA (no shfl sum-reduction); only the
// running row-max needs lane reductions (4 bpermute per row per 64 keys).
// ---------------------------------------------------------------------------
template <int NC, int DVC, int DH>
__global__ __launch_bounds__(128)
void mab_attn_kernel(const float* __restrict__ Qp,    // [B,N,DV] f32 (residual)
                     const bf16_t* __restrict__ Qpb,  // [B,N,DV] bf16
                     const bf16_t* __restrict__ Kp,   // [B,N,DV] bf16
                     const bf16_t* __restrict__ Vt,   // [B,DV,N] bf16 (transposed)
                     float* __restrict__ Oattn,       // [B,N,DV] f32
                     float scale) {
  __shared__ bf16_t plds[4][16 * 64];

  const int lane = threadIdx.x & 31;
  const int wave = threadIdx.x >> 5;
  const int q0 = blockIdx.x * 64 + wave * 16;
  const int h  = blockIdx.y;
  const int b  = blockIdx.z;
  const int c0 = h * DH;

  const float*  Qb  = Qp  + (size_t)b * NC * DVC;
  const bf16_t* Qbb = Qpb + (size_t)b * NC * DVC;
  const bf16_t* Kb  = Kp  + (size_t)b * NC * DVC;
  const bf16_t* Vb  = Vt  + (size_t)b * DVC * NC;

  // Q A-fragments (two 32-wide d chunks) held in VGPRs for the whole loop
  v16bf qa0 = load_A_frag_bf(Qbb + (size_t)q0 * DVC + c0 + 0,  DVC, lane);
  v16bf qa1 = load_A_frag_bf(Qbb + (size_t)q0 * DVC + c0 + 32, DVC, lane);

  // all-ones B fragment: rowsum(P) = P @ 1
  v16bf onesb;
#pragma unroll
  for (int i = 0; i < 16; ++i) onesb[i] = (__bf16)1.0f;

  float mrow[8];
#pragma unroll
  for (int i = 0; i < 8; ++i) mrow[i] = -3.0e38f;
  v8f acc[4] = {};
  v8f accl = {}; // running softmax denominator per row (replicated over lanes)

  const int colL   = lane & 15;
  const int rowAdd = (lane < 16) ? 0 : 8;
  bf16_t* pl = &plds[wave][0];

  for (int kb = 0; kb < NC; kb += 64) {
    if (kb + 64 < NC)
      __builtin_prefetch(Kb + (size_t)(kb + 64) * DVC + c0, 0, 3);

    // S = Q Kh^T : four 16x16 key tiles (K-fragments die after these WMMAs)
    v8f s[4];
#pragma unroll
    for (int st = 0; st < 4; ++st) {
      const bf16_t* kbase = Kb + (size_t)(kb + st * 16) * DVC + c0;
      v16bf k0 = load_B_frag_bf(kbase + 0,  DVC, lane); // d 0..31
      v16bf k1 = load_B_frag_bf(kbase + 32, DVC, lane); // d 32..63
      v8f z = {};
      z = wmma_bf16(qa0, k0, z);
      z = wmma_bf16(qa1, k1, z);
      s[st] = z;
    }
#pragma unroll
    for (int st = 0; st < 4; ++st)
#pragma unroll
      for (int i = 0; i < 8; ++i) s[st][i] *= scale;

    // running row-max (only remaining lane reduction; stays in 16-lane halves)
    float al[8];
#pragma unroll
    for (int i = 0; i < 8; ++i) {
      float r = fmaxf(fmaxf(s[0][i], s[1][i]), fmaxf(s[2][i], s[3][i]));
      r = fmaxf(r, __shfl_xor(r, 1));
      r = fmaxf(r, __shfl_xor(r, 2));
      r = fmaxf(r, __shfl_xor(r, 4));
      r = fmaxf(r, __shfl_xor(r, 8));
      float mnew = fmaxf(mrow[i], r);
      al[i] = __expf(mrow[i] - mnew);
      mrow[i] = mnew;
    }
#pragma unroll
    for (int st = 0; st < 4; ++st)
#pragma unroll
      for (int i = 0; i < 8; ++i) s[st][i] = __expf(s[st][i] - mrow[i]);
#pragma unroll
    for (int nt = 0; nt < 4; ++nt)
#pragma unroll
      for (int i = 0; i < 8; ++i) acc[nt][i] *= al[i];
#pragma unroll
    for (int i = 0; i < 8; ++i) accl[i] *= al[i];

    // V fragments: issued here so their latency overlaps the LDS round trip
    v16bf vb[4][2];
#pragma unroll
    for (int nt = 0; nt < 4; ++nt) {
      vb[nt][0] = load_B_frag_bf(Vb + (size_t)(c0 + nt * 16) * NC + kb, NC, lane);
      vb[nt][1] = load_B_frag_bf(Vb + (size_t)(c0 + nt * 16) * NC + kb + 32, NC, lane);
    }

    // P (16x64, C layout) -> LDS -> two A fragments (per-wave region)
#pragma unroll
    for (int st = 0; st < 4; ++st)
#pragma unroll
      for (int i = 0; i < 8; ++i)
        pl[(rowAdd + i) * 64 + st * 16 + colL] = f2b(s[st][i]);
    asm volatile("" ::: "memory");
    v16bf pa0 = load_A_frag_bf(pl + 0,  64, lane);
    v16bf pa1 = load_A_frag_bf(pl + 32, 64, lane);
    asm volatile("" ::: "memory");

    // O += P @ V ; denominator += P @ 1
#pragma unroll
    for (int nt = 0; nt < 4; ++nt) {
      acc[nt] = wmma_bf16(pa0, vb[nt][0], acc[nt]);
      acc[nt] = wmma_bf16(pa1, vb[nt][1], acc[nt]);
    }
    accl = wmma_bf16(pa0, onesb, accl);
    accl = wmma_bf16(pa1, onesb, accl);
  }

  // epilogue: normalize + Qp residual
  float invl[8];
#pragma unroll
  for (int i = 0; i < 8; ++i) invl[i] = 1.0f / accl[i];
#pragma unroll
  for (int nt = 0; nt < 4; ++nt) {
#pragma unroll
    for (int i = 0; i < 8; ++i) {
      int row = q0 + rowAdd + i;
      int col = c0 + nt * 16 + colL;
      float v = acc[nt][i] * invl[i] + Qb[(size_t)row * DVC + col];
      Oattn[((size_t)b * NC + row) * DVC + col] = v;
    }
  }
}

// ---------------------------------------------------------------------------
// LayerNorm: one D-channel row per D-thread block; optional bf16 copy out
// ---------------------------------------------------------------------------
template <int D>
__global__ __launch_bounds__(D)
void mab_ln_kernel(const float* __restrict__ X, const float* __restrict__ g,
                   const float* __restrict__ beta, float* __restrict__ Y,
                   bf16_t* __restrict__ Yb) {
  __shared__ float rs[D / 32], rs2[D / 32];
  const int row = blockIdx.x;
  const int t = threadIdx.x;
  const int lane = t & 31, wv = t >> 5;

  float x = X[(size_t)row * D + t];
  float s = x, s2 = x * x;
#pragma unroll
  for (int off = 16; off >= 1; off >>= 1) {
    s  += __shfl_xor(s, off);
    s2 += __shfl_xor(s2, off);
  }
  if (lane == 0) { rs[wv] = s; rs2[wv] = s2; }
  __syncthreads();
  if (wv == 0) {
    float a  = (lane < D / 32) ? rs[lane]  : 0.0f;
    float a2 = (lane < D / 32) ? rs2[lane] : 0.0f;
#pragma unroll
    for (int off = 4; off >= 1; off >>= 1) {
      a  += __shfl_xor(a, off);
      a2 += __shfl_xor(a2, off);
    }
    if (lane == 0) { rs[0] = a; rs2[0] = a2; }
  }
  __syncthreads();
  float mean = rs[0] / (float)D;
  float var  = rs2[0] / (float)D - mean * mean;
  float inv  = rsqrtf(var + 1e-5f);
  float y = (x - mean) * inv * g[t] + beta[t];
  size_t idx = (size_t)row * D + t;
  if (Y)  Y[idx] = y;
  if (Yb) Yb[idx] = f2b(y);
}

// ---------------------------------------------------------------------------
// Launcher
// ---------------------------------------------------------------------------
extern "C" void kernel_launch(void* const* d_in, const int* in_sizes, int n_in,
                              void* d_out, int out_size, void* d_ws, size_t ws_size,
                              hipStream_t stream) {
  const int Bc = 4, Nc = 2048, Dc = 256, Hc = 4;
  const int Mtok = Bc * Nc;    // 8192
  const int Ntok = Mtok * Dc;  // 2,097,152
  const int Nwt  = Dc * Dc;    // 65,536

  const float* Q   = (const float*)d_in[0];
  const float* Kin = (const float*)d_in[1];
  const float* Wq  = (const float*)d_in[2];
  const float* bq  = (const float*)d_in[3];
  const float* Wk  = (const float*)d_in[4];
  const float* bk  = (const float*)d_in[5];
  const float* Wv  = (const float*)d_in[6];
  const float* bv  = (const float*)d_in[7];
  const float* W0  = (const float*)d_in[8];
  const float* b0  = (const float*)d_in[9];
  const float* W1  = (const float*)d_in[10];
  const float* b1  = (const float*)d_in[11];
  const float* ln0g = (const float*)d_in[12];
  const float* ln0b = (const float*)d_in[13];
  const float* ln1g = (const float*)d_in[14];
  const float* ln1b = (const float*)d_in[15];

  char* ws = (char*)d_ws;
  size_t off = 0;
  bf16_t* Qbf = (bf16_t*)(ws + off); off += (size_t)Ntok * 2;
  bf16_t* Kbf = (bf16_t*)(ws + off); off += (size_t)Ntok * 2;
  bf16_t* Wqb = (bf16_t*)(ws + off); off += (size_t)Nwt * 2;
  bf16_t* Wkb = (bf16_t*)(ws + off); off += (size_t)Nwt * 2;
  bf16_t* Wvb = (bf16_t*)(ws + off); off += (size_t)Nwt * 2;
  bf16_t* W0b = (bf16_t*)(ws + off); off += (size_t)Nwt * 2;
  bf16_t* W1b = (bf16_t*)(ws + off); off += (size_t)Nwt * 2;
  float*  Qp  = (float*)(ws + off);  off += (size_t)Ntok * 4;
  bf16_t* Qpb = (bf16_t*)(ws + off); off += (size_t)Ntok * 2;
  bf16_t* Kpb = (bf16_t*)(ws + off); off += (size_t)Ntok * 2;
  bf16_t* Vtb = (bf16_t*)(ws + off); off += (size_t)Ntok * 2;
  float*  Oa  = (float*)(ws + off);  off += (size_t)Ntok * 4;  // reused for X2
  float*  X0  = (float*)(ws + off);  off += (size_t)Ntok * 4;
  bf16_t* X0b = (bf16_t*)(ws + off); off += (size_t)Ntok * 2;
  bf16_t* H1b = (bf16_t*)(ws + off); off += (size_t)Ntok * 2;
  float*  X2  = Oa; // Oa is dead after LN0

  // --- one-time f32 -> bf16 conversions (cold, bandwidth-trivial) ---
  dim3 bC(256);
  mab_cvt_bf16_kernel<<<dim3((Ntok / 4 + 255) / 256), bC, 0, stream>>>(Q,   Qbf, Ntok);
  mab_cvt_bf16_kernel<<<dim3((Ntok / 4 + 255) / 256), bC, 0, stream>>>(Kin, Kbf, Ntok);
  mab_cvt_bf16_kernel<<<dim3((Nwt / 4 + 255) / 256),  bC, 0, stream>>>(Wq, Wqb, Nwt);
  mab_cvt_bf16_kernel<<<dim3((Nwt / 4 + 255) / 256),  bC, 0, stream>>>(Wk, Wkb, Nwt);
  mab_cvt_bf16_kernel<<<dim3((Nwt / 4 + 255) / 256),  bC, 0, stream>>>(Wv, Wvb, Nwt);
  mab_cvt_bf16_kernel<<<dim3((Nwt / 4 + 255) / 256),  bC, 0, stream>>>(W0, W0b, Nwt);
  mab_cvt_bf16_kernel<<<dim3((Nwt / 4 + 255) / 256),  bC, 0, stream>>>(W1, W1b, Nwt);

  dim3 gG(Mtok / 128, Dc / 64), bG(128);
  // Qp = Q @ Wq^T + bq (f32 for residual + bf16 for score WMMA)
  mab_gemm_kernel<256, 256, 2048><<<gG, bG, 0, stream>>>(Qbf, Wqb, bq, nullptr, Qp,
                                                         Qpb, nullptr, 0);
  // Kp = K @ Wk^T + bk (bf16 row-major)
  mab_gemm_kernel<256, 256, 2048><<<gG, bG, 0, stream>>>(Kbf, Wkb, bk, nullptr,
                                                         nullptr, Kpb, nullptr, 0);
  // Vp = K @ Wv^T + bv (bf16, transposed [B,DV,N])
  mab_gemm_kernel<256, 256, 2048><<<gG, bG, 0, stream>>>(Kbf, Wvb, bv, nullptr,
                                                         nullptr, nullptr, Vtb, 0);
  // attention + residual
  dim3 gA(Nc / 64, Hc, Bc), bA(128);
  mab_attn_kernel<2048, 256, 64><<<gA, bA, 0, stream>>>(Qp, Qpb, Kpb, Vtb, Oa,
                                                        1.0f / 16.0f);
  // LN0 -> X0 (f32 residual) + X0b (bf16 GEMM input)
  mab_ln_kernel<256><<<dim3(Mtok), dim3(Dc), 0, stream>>>(Oa, ln0g, ln0b, X0, X0b);
  // H1 = relu(X0 @ W0^T + b0)  (bf16 only; feeds next GEMM)
  mab_gemm_kernel<256, 256, 2048><<<gG, bG, 0, stream>>>(X0b, W0b, b0, nullptr,
                                                         nullptr, H1b, nullptr, 1);
  // X2 = X0 + relu(H1 @ W1^T + b1)
  mab_gemm_kernel<256, 256, 2048><<<gG, bG, 0, stream>>>(H1b, W1b, b1, X0, X2,
                                                         nullptr, nullptr, 1);
  // LN1 -> out
  mab_ln_kernel<256><<<dim3(Mtok), dim3(Dc), 0, stream>>>(X2, ln1g, ln1b,
                                                          (float*)d_out, nullptr);
}
